// HeterogeneousGNNModel_17343077941394
// MI455X (gfx1250) — matrix operand order repaired and checked by reading, not compiled, not run on hardware
//
#include <hip/hip_runtime.h>
#include <math.h>

// ---------------------------------------------------------------------------
// HeterogeneousGNN (2-layer GATv2 + MLP head) for MI455X / gfx1250, wave32.
// Dense projections: V_WMMA_F32_16X16X32_F16, one wave per 16xN row stripe,
// A and B fragments loaded as 2x b128 each (weights pre-transposed to [N][K]).
// Edge softmax/aggregation: float atomics (int-ordering max trick + add).
// c2_ta conv is skipped: a2 is dead w.r.t. the output head.
//
// Input order (setup_inputs dict order, params flattened in insertion order):
//  0 x_tag[50000*64] 1 x_ant[128*64]
//  2 tt_src 3 tt_dst 4 ta_src 5 ta_dst 6 at_src 7 at_dst
//  8 ea_tt[800000] 9 ea_ta[400000] 10 ea_at[400000]
//  gat params {Wl,Wr,We,att,b}: 11..15 c1_tt, 16..20 c1_ta, 21..25 c1_at,
//                               26..30 c2_tt, 31..35 c2_ta, 36..40 c2_at
//  41 res_W[64*96] 42 res_b[96]
//  43 fc_W1[96*32] 44 fc_b1[32] 45 fc_W2[32*32] 46 fc_b2[32]
//  47 fc_W3[32*2] 48 fc_b3[2]
// ---------------------------------------------------------------------------

#define NT      50000
#define NA      128
#define DIN     64
#define HCW     96          // heads*ch, same for both layers
#define ETT     800000LL
#define ETA     400000LL
#define EAT     400000LL
#define NEG_SLOPE 0.2f

typedef __attribute__((ext_vector_type(16))) _Float16 v16h;
typedef __attribute__((ext_vector_type(8)))  _Float16 v8h;
typedef __attribute__((ext_vector_type(8)))  float    v8f;
typedef __attribute__((ext_vector_type(4)))  float    f4;

#define GS_LOOP(i, n)                                                          \
  for (long long i = blockIdx.x * (long long)blockDim.x + threadIdx.x;         \
       i < (n); i += (long long)gridDim.x * blockDim.x)

// ------------------------------ WMMA GEMM ----------------------------------
// C[M,N] = A[M,K](f16) * W[K,N] [+ bias] [relu], f32 out.
// W supplied TRANSPOSED as Wt[N,K] so each lane's B fragment is contiguous.
// One wave per 16-row stripe covering all NTILES 16-col tiles; 4 waves/block.
// M multiple of 16, N == 16*NTILES, K multiple of 32. All rows 16B-aligned.
template <int NTILES>
__global__ __launch_bounds__(128) void gemm_wmma_k(
    const _Float16* __restrict__ A, const _Float16* __restrict__ Wt,
    const float* __restrict__ bias, float* __restrict__ C,
    int M, int N, int K, int relu)
{
  const int lane = threadIdx.x & 31;
  const int wave = threadIdx.x >> 5;
  const int mt = blockIdx.x * 4 + wave;
  if (mt * 16 >= M) return;                 // wave-uniform; EXEC stays all-1s
  const int r16 = lane & 15;
  const int hi  = lane >> 4;                // lane half select
  const _Float16* Arow = A + (size_t)(mt * 16 + r16) * K;
  v8f acc[NTILES];
#pragma unroll
  for (int t = 0; t < NTILES; ++t) acc[t] = (v8f){};

  for (int k0 = 0; k0 < K; k0 += 32) {
    // A 16x32 f16 fragment (ISA 7.12.2): lane holds row r16,
    // halves 0..7 = K[k0+hi*8 ..], halves 8..15 = K[k0+16+hi*8 ..] (contiguous)
    const v8h alo = *(const v8h*)(Arow + k0 + hi * 8);
    const v8h ahi = *(const v8h*)(Arow + k0 + 16 + hi * 8);
    const v16h a = __builtin_shufflevector(alo, ahi, 0, 1, 2, 3, 4, 5, 6, 7,
                                           8, 9, 10, 11, 12, 13, 14, 15);
#pragma unroll
    for (int t = 0; t < NTILES; ++t) {
      // B 32x16 fragment: lane holds col t*16+r16, K = k0 + hi*8 .. +7 and
      // k0 + 16 + hi*8 .. +7 (same order as A), from transposed weight Wt[N][K].
      const _Float16* Wrow = Wt + (size_t)(t * 16 + r16) * K + k0 + hi * 8;
      const v8h blo = *(const v8h*)(Wrow);
      const v8h bhi = *(const v8h*)(Wrow + 16);
      const v16h b = __builtin_shufflevector(blo, bhi, 0, 1, 2, 3, 4, 5, 6, 7,
                                             8, 9, 10, 11, 12, 13, 14, 15);
      acc[t] = __builtin_amdgcn_wmma_f32_16x16x32_f16(false, a, false, b,
                                                      (short)0, acc[t], false, false);
    }
  }
#pragma unroll
  for (int t = 0; t < NTILES; ++t) {
    const int n = t * 16 + r16;
    const float bv = bias ? bias[n] : 0.0f;
#pragma unroll
    for (int r = 0; r < 8; ++r) {           // C/D: VGPR r -> M = r + 8*hi
      float v = acc[t][r] + bv;
      if (relu && v < 0.0f) v = 0.0f;
      C[(size_t)(mt * 16 + r + 8 * hi) * N + n] = v;
    }
  }
}

// ---------------------------- utility kernels ------------------------------
__global__ void fill_k(float* __restrict__ p, float v, long long n) {
  GS_LOOP(i, n) p[i] = v;
}

__global__ void f32_to_f16_k(const float* __restrict__ x, _Float16* __restrict__ y,
                             long long n) {
  GS_LOOP(i, n) y[i] = (_Float16)x[i];
}

// Fused convert + transpose: W[K][N] f32 -> Wt[N][K] f16.
__global__ void cvt_tr_k(const float* __restrict__ W, _Float16* __restrict__ Wt,
                         int K, int N) {
  int i = blockIdx.x * blockDim.x + threadIdx.x;
  if (i < K * N) {
    int k = i / N, n = i - k * N;
    Wt[(size_t)n * K + k] = (_Float16)W[i];
  }
}

__global__ void mean_k(const float* __restrict__ x, float* __restrict__ out,
                       long long n) {
  const float inv = 1.0f / (float)n;
  GS_LOOP(i, n) atomicAdd(out, x[i] * inv);
}

__global__ void vec_add3_k(float* __restrict__ d, const float* __restrict__ a,
                           const float* __restrict__ b, const float* __restrict__ c,
                           int n) {
  int i = blockIdx.x * blockDim.x + threadIdx.x;
  if (i < n) d[i] = a[i] + b[i] + (c ? c[i] : 0.0f);
}

__global__ void rows_bias_k(float* __restrict__ d, const float* __restrict__ x,
                            const float* __restrict__ bias, long long n, int w) {
  GS_LOOP(i, n) d[i] = x[i] + bias[(int)(i % w)];
}

__global__ void elu_k(float* __restrict__ p, long long n) {
  GS_LOOP(i, n) { float v = p[i]; p[i] = v > 0.0f ? v : expm1f(v); }
}

// ------------------------------ edge kernels -------------------------------
__device__ __forceinline__ void atomicMaxF(float* addr, float v) {
  if (v >= 0.0f) atomicMax((int*)addr, __float_as_int(v));
  else           atomicMin((unsigned int*)addr, __float_as_uint(v));
}

// One thread per (edge, head): leaky-relu'd attention score + segment max.
// src==nullptr -> self-loop mode (src=dst=e, ea = *eamean).
__global__ void edge_score_k(const int* __restrict__ src, const int* __restrict__ dst,
                             const float* __restrict__ ea, const float* __restrict__ eamean,
                             const float* __restrict__ xl, const float* __restrict__ xr,
                             const float* __restrict__ We, const float* __restrict__ att,
                             float* __restrict__ score, float* __restrict__ smax,
                             long long E, long long scoreOff, int heads, int ch) {
  const int hc = heads * ch;
  GS_LOOP(idx, E * heads) {
    const long long e = idx / heads;
    const int h = (int)(idx - e * heads);
    const int s = src ? src[e] : (int)e;
    const int d = dst ? dst[e] : (int)e;
    const float eav = ea ? ea[e] : eamean[0];
    const float* pl = xl + (size_t)s * hc + h * ch;
    const float* pr = xr + (size_t)d * hc + h * ch;
    const float* pw = We + h * ch;
    const float* pa = att + h * ch;
    float sc = 0.0f;
    for (int c = 0; c < ch; c += 4) {       // 16B-aligned: ch%4==0, hc=96
      const f4 l = *(const f4*)(pl + c);
      const f4 r = *(const f4*)(pr + c);
      const f4 w = *(const f4*)(pw + c);
      const f4 t = *(const f4*)(pa + c);
#pragma unroll
      for (int u = 0; u < 4; ++u) {
        float m = l[u] + r[u] + eav * w[u];
        m = m > 0.0f ? m : NEG_SLOPE * m;   // leaky_relu(0.2)
        sc += m * t[u];
      }
    }
    score[(scoreOff + e) * heads + h] = sc;
    atomicMaxF(&smax[(size_t)d * heads + h], sc);
  }
}

// One thread per (edge, head): exp(score - smax) in place + segment sum.
__global__ void edge_exp_k(const int* __restrict__ dst, float* __restrict__ score,
                           const float* __restrict__ smax, float* __restrict__ den,
                           long long E, long long scoreOff, int heads) {
  GS_LOOP(idx, E * heads) {
    const long long e = idx / heads;
    const int h = (int)(idx - e * heads);
    const int d = dst ? dst[e] : (int)e;
    float sm = smax[(size_t)d * heads + h];
    if (!(sm >= -3.402823466e38f && sm <= 3.402823466e38f)) sm = 0.0f; // !finite
    const long long si = (scoreOff + e) * heads + h;
    const float ex = expf(score[si] - sm);
    score[si] = ex;
    atomicAdd(&den[(size_t)d * heads + h], ex);
  }
}

// One thread per (edge, 4 channels): out[dst] += alpha * xl[src].
__global__ void edge_aggr_k(const int* __restrict__ src, const int* __restrict__ dst,
                            const float* __restrict__ score, const float* __restrict__ den,
                            const float* __restrict__ xl, float* __restrict__ out,
                            long long E, long long scoreOff, int heads, int ch) {
  const int hc = heads * ch;
  const int q = hc >> 2;                    // channel quads per edge
  GS_LOOP(idx, E * q) {
    const long long e = idx / q;
    const int j4 = (int)(idx - e * q) << 2; // first channel of the quad
    const int h = j4 / ch;                  // quad never straddles heads (ch%4==0)
    const int s = src ? src[e] : (int)e;
    const int d = dst ? dst[e] : (int)e;
    const float ex = score[(scoreOff + e) * heads + h];
    const float alpha = ex / (den[(size_t)d * heads + h] + 1e-16f);
    const f4 xv = *(const f4*)(xl + (size_t)s * hc + j4);
    float* op = out + (size_t)d * hc + j4;
    atomicAdd(op + 0, alpha * xv[0]);
    atomicAdd(op + 1, alpha * xv[1]);
    atomicAdd(op + 2, alpha * xv[2]);
    atomicAdd(op + 3, alpha * xv[3]);
  }
}

// Final 32 -> 2 head (too skinny for WMMA; VALU dot products).
__global__ void fc3_k(const float* __restrict__ h, const float* __restrict__ W,
                      const float* __restrict__ b, float* __restrict__ out, int M) {
  int i = blockIdx.x * blockDim.x + threadIdx.x;
  if (i >= M) return;
  float a0 = b[0], a1 = b[1];
  const float* hp = h + (size_t)i * 32;
#pragma unroll
  for (int c = 0; c < 32; ++c) { a0 += hp[c] * W[c * 2]; a1 += hp[c] * W[c * 2 + 1]; }
  out[(size_t)i * 2]     = a0;
  out[(size_t)i * 2 + 1] = a1;
}

// ------------------------------ host helpers -------------------------------
static inline int gblocks(long long n, int tpb) {
  long long b = (n + tpb - 1) / tpb;
  if (b > 1048576) b = 1048576;
  if (b < 1) b = 1;
  return (int)b;
}

static inline void* ws_take(char*& p, size_t bytes) {
  void* r = (void*)p;
  p += (bytes + 255) & ~(size_t)255;
  return r;
}

static void gemm(hipStream_t st, const _Float16* A, const _Float16* Wt,
                 const float* bias, float* C, int M, int N, int K, int relu) {
  const int mtiles = (M + 15) / 16;
  dim3 grid((mtiles + 3) / 4);
  if (N == 96)
    gemm_wmma_k<6><<<grid, 128, 0, st>>>(A, Wt, bias, C, M, N, K, relu);
  else
    gemm_wmma_k<2><<<grid, 128, 0, st>>>(A, Wt, bias, C, M, N, K, relu);
}

static void run_conv(hipStream_t st,
                     const int* src, const int* dst, const float* ea, const float* eamean,
                     const float* xl, const float* xr,
                     const float* We, const float* att,
                     float* score, float* smax, float* den, float* out,
                     long long E, long long nSrc, long long nDst,
                     int heads, int ch, int selfLoops) {
  const long long seg = nDst * heads;
  const long long q = (long long)(heads * ch) >> 2;
  fill_k<<<gblocks(seg, 256), 256, 0, st>>>(smax, -INFINITY, seg);
  fill_k<<<gblocks(seg, 256), 256, 0, st>>>(den, 0.0f, seg);
  edge_score_k<<<gblocks(E * heads, 256), 256, 0, st>>>(
      src, dst, ea, nullptr, xl, xr, We, att, score, smax, E, 0, heads, ch);
  if (selfLoops)
    edge_score_k<<<gblocks(nSrc * heads, 256), 256, 0, st>>>(
        nullptr, nullptr, nullptr, eamean, xl, xr, We, att, score, smax, nSrc, E, heads, ch);
  edge_exp_k<<<gblocks(E * heads, 256), 256, 0, st>>>(dst, score, smax, den, E, 0, heads);
  if (selfLoops)
    edge_exp_k<<<gblocks(nSrc * heads, 256), 256, 0, st>>>(nullptr, score, smax, den, nSrc, E, heads);
  edge_aggr_k<<<gblocks(E * q, 256), 256, 0, st>>>(
      src, dst, score, den, xl, out, E, 0, heads, ch);
  if (selfLoops)
    edge_aggr_k<<<gblocks(nSrc * q, 256), 256, 0, st>>>(
        nullptr, nullptr, score, den, xl, out, nSrc, E, heads, ch);
}

// ------------------------------ orchestration ------------------------------
extern "C" void kernel_launch(void* const* d_in, const int* in_sizes, int n_in,
                              void* d_out, int out_size, void* d_ws, size_t ws_size,
                              hipStream_t stream) {
  (void)in_sizes; (void)n_in; (void)out_size; (void)ws_size;
  const float* x_tag  = (const float*)d_in[0];
  const float* x_ant  = (const float*)d_in[1];
  const int*   tt_src = (const int*)d_in[2];
  const int*   tt_dst = (const int*)d_in[3];
  const int*   ta_src = (const int*)d_in[4];
  const int*   ta_dst = (const int*)d_in[5];
  const int*   at_src = (const int*)d_in[6];
  const int*   at_dst = (const int*)d_in[7];
  const float* ea_tt  = (const float*)d_in[8];
  const float* ea_ta  = (const float*)d_in[9];
  const float* ea_at  = (const float*)d_in[10];
  // GAT params: base + {0:Wl,1:Wr,2:We,3:att,4:b}
  const float* Wl1tt = (const float*)d_in[11]; const float* Wr1tt = (const float*)d_in[12];
  const float* We1tt = (const float*)d_in[13]; const float* at1tt = (const float*)d_in[14];
  const float* b1tt  = (const float*)d_in[15];
  const float* Wl1ta = (const float*)d_in[16]; const float* Wr1ta = (const float*)d_in[17];
  const float* We1ta = (const float*)d_in[18]; const float* at1ta = (const float*)d_in[19];
  const float* b1ta  = (const float*)d_in[20];
  const float* Wl1at = (const float*)d_in[21]; const float* Wr1at = (const float*)d_in[22];
  const float* We1at = (const float*)d_in[23]; const float* at1at = (const float*)d_in[24];
  const float* b1at  = (const float*)d_in[25];
  const float* Wl2tt = (const float*)d_in[26]; const float* Wr2tt = (const float*)d_in[27];
  const float* We2tt = (const float*)d_in[28]; const float* at2tt = (const float*)d_in[29];
  const float* b2tt  = (const float*)d_in[30];
  // d_in[31..35] = c2_ta : dead code (a2 unused by output head) -> skipped
  const float* Wl2at = (const float*)d_in[36]; const float* Wr2at = (const float*)d_in[37];
  const float* We2at = (const float*)d_in[38]; const float* at2at = (const float*)d_in[39];
  const float* b2at  = (const float*)d_in[40];
  const float* res_W = (const float*)d_in[41]; const float* res_b = (const float*)d_in[42];
  const float* fc_W1 = (const float*)d_in[43]; const float* fc_b1 = (const float*)d_in[44];
  const float* fc_W2 = (const float*)d_in[45]; const float* fc_b2 = (const float*)d_in[46];
  const float* fc_W3 = (const float*)d_in[47]; const float* fc_b3 = (const float*)d_in[48];
  float* out = (float*)d_out;

  // ---------------- workspace layout (~160 MB) ----------------
  char* wp = (char*)d_ws;
  _Float16* x_tag_h = (_Float16*)ws_take(wp, (size_t)NT * DIN * 2);
  _Float16* x_ant_h = (_Float16*)ws_take(wp, (size_t)NA * DIN * 2);
  _Float16* t_h     = (_Float16*)ws_take(wp, (size_t)NT * HCW * 2);  // t1 then t2
  _Float16* a1_h    = (_Float16*)ws_take(wp, (size_t)NA * HCW * 2);
  _Float16* h1_h    = (_Float16*)ws_take(wp, (size_t)NT * 32 * 2);
  // f16 weights, TRANSPOSED to [N][K]
  _Float16* wl1tt_h = (_Float16*)ws_take(wp, (size_t)DIN * HCW * 2);
  _Float16* wr1tt_h = (_Float16*)ws_take(wp, (size_t)DIN * HCW * 2);
  _Float16* wl1ta_h = (_Float16*)ws_take(wp, (size_t)DIN * HCW * 2);
  _Float16* wr1ta_h = (_Float16*)ws_take(wp, (size_t)DIN * HCW * 2);
  _Float16* wl1at_h = (_Float16*)ws_take(wp, (size_t)DIN * HCW * 2);
  _Float16* wr1at_h = (_Float16*)ws_take(wp, (size_t)DIN * HCW * 2);
  _Float16* resW_h  = (_Float16*)ws_take(wp, (size_t)DIN * HCW * 2);
  _Float16* wl2tt_h = (_Float16*)ws_take(wp, (size_t)HCW * HCW * 2);
  _Float16* wr2tt_h = (_Float16*)ws_take(wp, (size_t)HCW * HCW * 2);
  _Float16* wl2at_h = (_Float16*)ws_take(wp, (size_t)HCW * HCW * 2);
  _Float16* wr2at_h = (_Float16*)ws_take(wp, (size_t)HCW * HCW * 2);
  _Float16* fcW1_h  = (_Float16*)ws_take(wp, (size_t)HCW * 32 * 2);
  _Float16* fcW2_h  = (_Float16*)ws_take(wp, (size_t)32 * 32 * 2);
  // f32 working buffers
  float* xlA    = (float*)ws_take(wp, (size_t)NT * HCW * 4);  // xl_tt / xl2_tt
  float* xrA    = (float*)ws_take(wp, (size_t)NT * HCW * 4);  // xr_tt / xr2_tt
  float* xrB    = (float*)ws_take(wp, (size_t)NT * HCW * 4);  // xr_at / xr2_at
  float* xlC    = (float*)ws_take(wp, (size_t)NT * HCW * 4);  // xl_ta (L1 only)
  float* sxl_at = (float*)ws_take(wp, (size_t)NA * HCW * 4);  // ant-side xl (L1 at / L2 at)
  float* sxr_ta = (float*)ws_take(wp, (size_t)NA * HCW * 4);  // ant-side xr (L1 ta)
  float* tacc   = (float*)ws_take(wp, (size_t)NT * HCW * 4);  // -> t1
  float* t2acc  = (float*)ws_take(wp, (size_t)NT * HCW * 4);  // -> t2
  float* aacc   = (float*)ws_take(wp, (size_t)NA * HCW * 4);  // -> a1
  float* score  = (float*)ws_take(wp, (size_t)(ETT + NT) * 3 * 4);
  float* smax   = (float*)ws_take(wp, (size_t)NT * 3 * 4);
  float* den    = (float*)ws_take(wp, (size_t)NT * 3 * 4);
  float* h1     = (float*)ws_take(wp, (size_t)NT * 32 * 4);
  float* h2     = (float*)ws_take(wp, (size_t)NT * 32 * 4);
  float* eamean = (float*)ws_take(wp, 256);
  float* cb_t   = (float*)ws_take(wp, HCW * 4);
  float* cb_a   = (float*)ws_take(wp, HCW * 4);
  float* bsum2  = (float*)ws_take(wp, HCW * 4);

  // ---------------- conversions ----------------
  // Activations: plain f32 -> f16.
  f32_to_f16_k<<<gblocks((long long)NT * DIN, 256), 256, 0, stream>>>(
      x_tag, x_tag_h, (long long)NT * DIN);
  f32_to_f16_k<<<gblocks((long long)NA * DIN, 256), 256, 0, stream>>>(
      x_ant, x_ant_h, (long long)NA * DIN);
  // Weights: fused convert + transpose to [N][K].
  struct WCvt { const float* s; _Float16* d; int K, N; };
  const WCvt wv[] = {
    {Wl1tt, wl1tt_h, DIN, HCW}, {Wr1tt, wr1tt_h, DIN, HCW},
    {Wl1ta, wl1ta_h, DIN, HCW}, {Wr1ta, wr1ta_h, DIN, HCW},
    {Wl1at, wl1at_h, DIN, HCW}, {Wr1at, wr1at_h, DIN, HCW},
    {res_W, resW_h,  DIN, HCW},
    {Wl2tt, wl2tt_h, HCW, HCW}, {Wr2tt, wr2tt_h, HCW, HCW},
    {Wl2at, wl2at_h, HCW, HCW}, {Wr2at, wr2at_h, HCW, HCW},
    {fc_W1, fcW1_h,  HCW, 32},  {fc_W2, fcW2_h,  32, 32},
  };
  for (const WCvt& w : wv)
    cvt_tr_k<<<gblocks((long long)w.K * w.N, 256), 256, 0, stream>>>(w.s, w.d, w.K, w.N);

  // Combined biases + mean(ea_tt) for self-loop fill.
  vec_add3_k<<<1, HCW, 0, stream>>>(cb_t, res_b, b1tt, b1at, HCW);
  vec_add3_k<<<1, HCW, 0, stream>>>(cb_a, res_b, b1ta, nullptr, HCW);
  vec_add3_k<<<1, HCW, 0, stream>>>(bsum2, b2tt, b2at, nullptr, HCW);
  fill_k<<<1, 1, 0, stream>>>(eamean, 0.0f, 1);
  mean_k<<<gblocks(ETT, 256), 256, 0, stream>>>(ea_tt, eamean, ETT);

  // ---------------- layer 1: projections (WMMA) ----------------
  gemm(stream, x_tag_h, wl1tt_h, nullptr, xlA, NT, HCW, DIN, 0);
  gemm(stream, x_tag_h, wr1tt_h, nullptr, xrA, NT, HCW, DIN, 0);
  gemm(stream, x_tag_h, wr1at_h, nullptr, xrB, NT, HCW, DIN, 0);
  gemm(stream, x_tag_h, wl1ta_h, nullptr, xlC, NT, HCW, DIN, 0);
  gemm(stream, x_ant_h, wl1at_h, nullptr, sxl_at, NA, HCW, DIN, 0);
  gemm(stream, x_ant_h, wr1ta_h, nullptr, sxr_ta, NA, HCW, DIN, 0);
  gemm(stream, x_tag_h, resW_h, cb_t, tacc, NT, HCW, DIN, 0);   // residual + biases
  gemm(stream, x_ant_h, resW_h, cb_a, aacc, NA, HCW, DIN, 0);

  // ---------------- layer 1: GATv2 convs (heads=3, ch=32) ----------------
  run_conv(stream, tt_src, tt_dst, ea_tt, eamean, xlA, xrA, We1tt, at1tt,
           score, smax, den, tacc, ETT, NT, NT, 3, 32, 1);
  run_conv(stream, at_src, at_dst, ea_at, nullptr, sxl_at, xrB, We1at, at1at,
           score, smax, den, tacc, EAT, NA, NT, 3, 32, 0);
  elu_k<<<gblocks((long long)NT * HCW, 256), 256, 0, stream>>>(tacc, (long long)NT * HCW); // t1
  run_conv(stream, ta_src, ta_dst, ea_ta, nullptr, xlC, sxr_ta, We1ta, at1ta,
           score, smax, den, aacc, ETA, NT, NA, 3, 32, 0);
  elu_k<<<gblocks((long long)NA * HCW, 256), 256, 0, stream>>>(aacc, (long long)NA * HCW); // a1

  f32_to_f16_k<<<gblocks((long long)NT * HCW, 256), 256, 0, stream>>>(tacc, t_h, (long long)NT * HCW);
  f32_to_f16_k<<<gblocks((long long)NA * HCW, 256), 256, 0, stream>>>(aacc, a1_h, (long long)NA * HCW);

  // ---------------- layer 2: projections (WMMA) ----------------
  gemm(stream, t_h,  wl2tt_h, nullptr, xlA, NT, HCW, HCW, 0);
  gemm(stream, t_h,  wr2tt_h, nullptr, xrA, NT, HCW, HCW, 0);
  gemm(stream, t_h,  wr2at_h, nullptr, xrB, NT, HCW, HCW, 0);
  gemm(stream, a1_h, wl2at_h, nullptr, sxl_at, NA, HCW, HCW, 0);
  rows_bias_k<<<gblocks((long long)NT * HCW, 256), 256, 0, stream>>>(
      t2acc, tacc, bsum2, (long long)NT * HCW, HCW);           // identity residual + biases

  // ---------------- layer 2: GATv2 convs (heads=1, ch=96) ----------------
  run_conv(stream, tt_src, tt_dst, ea_tt, eamean, xlA, xrA, We2tt, at2tt,
           score, smax, den, t2acc, ETT, NT, NT, 1, HCW, 1);
  run_conv(stream, at_src, at_dst, ea_at, nullptr, sxl_at, xrB, We2at, at2at,
           score, smax, den, t2acc, EAT, NA, NT, 1, HCW, 0);
  elu_k<<<gblocks((long long)NT * HCW, 256), 256, 0, stream>>>(t2acc, (long long)NT * HCW); // t2

  // ---------------- MLP head (WMMA for 96->32 and 32->32) ----------------
  f32_to_f16_k<<<gblocks((long long)NT * HCW, 256), 256, 0, stream>>>(t2acc, t_h, (long long)NT * HCW);
  gemm(stream, t_h, fcW1_h, fc_b1, h1, NT, 32, HCW, 1);        // relu fused
  f32_to_f16_k<<<gblocks((long long)NT * 32, 256), 256, 0, stream>>>(h1, h1_h, (long long)NT * 32);
  gemm(stream, h1_h, fcW2_h, fc_b2, h2, NT, 32, 32, 1);        // relu fused
  fc3_k<<<gblocks(NT, 256), 256, 0, stream>>>(h2, fc_W3, fc_b3, out, NT);
}
